// AttentionLayer_4097398800467
// MI455X (gfx1250) — matrix operand order repaired and checked by reading, not compiled, not run on hardware
//
#include <hip/hip_runtime.h>
#include <hip/hip_bf16.h>
#include <math.h>

// Problem constants (match reference): T=1024, B=8, D=512, W=32
#define T_DIM 1024
#define B_DIM 8
#define D_DIM 512
#define W_WIN 32
#define M_ROWS (T_DIM * B_DIM) // 8192 rows of X = inputs viewed as [T*B, D]

typedef __attribute__((ext_vector_type(16))) __bf16 v16bf;
typedef __attribute__((ext_vector_type(8)))  float  v8f;

__device__ __forceinline__ unsigned short f2bf(float f) {
  // round-to-nearest-even f32 -> bf16 (inputs are finite normals)
  unsigned int u = __float_as_uint(f);
  u += 0x7FFFu + ((u >> 16) & 1u);
  return (unsigned short)(u >> 16);
}

// ---------------------------------------------------------------------------
// Kernel 0: convert weight_W [512x512] f32 (row k, col n) into bf16 WMMA
// B-tile layout. Tile (kt,nt) covers K=[kt*32,kt*32+32), N=[nt*16,nt*16+16).
// This code's contiguous-K B layout: lane L holds N = L%16, K = (L/16)*16 + e, e=0..15.
// Stored so each lane's 16 bf16 are contiguous: tile-major, lane-major.
// ---------------------------------------------------------------------------
__global__ void __launch_bounds__(256)
attn_wconvert(const float* __restrict__ w, unsigned short* __restrict__ wt) {
  int tid  = blockIdx.x * blockDim.x + threadIdx.x;   // 0..16383
  int lane = tid & 31;
  int tile = tid >> 5;                                // 0..511 = kt*32+nt
  int kt   = tile >> 5;                               // 0..15
  int nt   = tile & 31;                               // 0..31
  int kbase = kt * 32 + ((lane >> 4) << 4);           // +0 or +16
  int n     = nt * 16 + (lane & 15);
  unsigned short* dst = wt + ((size_t)tile * 32 + lane) * 16;
  const float*    src = w + (size_t)kbase * D_DIM + n;
#pragma unroll
  for (int e = 0; e < 16; ++e)
    dst[e] = f2bf(src[(size_t)e * D_DIM]);
}

// ---------------------------------------------------------------------------
// Kernel 1: s[row] = tanh(x[row,:] @ W) . proj   for row in [0, 8192).
// One wave per 16-row M-tile. A (16x512, bf16) preloaded into 128 VGPRs in
// this code's contiguous-K A-layout (matches B):
//   lanes 0-15  (M=lane)   : u[0..15]=K0..15  (per 32-K tile)
//   lanes 16-31 (M=lane-16): u[0..15]=K16..31
// C layout: VGPR r -> row (half*8 + r), col = lane%16.
// amdgpu_waves_per_eu(1): this is a one-shot 512-wave kernel; occupancy is
// irrelevant, so give the wave the full register file and avoid the scratch
// spills of the A tiles observed at the default VGPR budget.
// ---------------------------------------------------------------------------
__global__ void __launch_bounds__(256) __attribute__((amdgpu_waves_per_eu(1)))
attn_scores(const float* __restrict__ x,
            const unsigned short* __restrict__ wt,
            const float* __restrict__ proj,
            float* __restrict__ s) {
  const int lane = threadIdx.x & 31;
  const int wave = threadIdx.x >> 5;
  const int half = lane >> 4;
  const int mloc = lane & 15;
  const int m0   = (blockIdx.x * 8 + wave) * 16;      // 64 blocks * 8 waves * 16 = 8192
  const float* xrow = x + (size_t)(m0 + mloc) * D_DIM;

  union { v16bf v; unsigned short u[16]; } a[16];
#pragma unroll
  for (int kt = 0; kt < 16; ++kt) {
    const float* p0 = xrow + kt * 32 + half * 16;     // K = kt*32 + half*16 + e
    const float* p1 = p0 + 8;                         // K = kt*32 + half*16 + 8 + e
#pragma unroll
    for (int e = 0; e < 8; ++e) {
      a[kt].u[e]     = f2bf(p0[e]);
      a[kt].u[8 + e] = f2bf(p1[e]);
    }
  }

  float rs[8];
#pragma unroll
  for (int r = 0; r < 8; ++r) rs[r] = 0.0f;

  for (int nt = 0; nt < 32; ++nt) {
    float pj = proj[nt * 16 + mloc];                  // hoisted epilogue weight
    v8f acc = {};
#pragma unroll
    for (int kt = 0; kt < 16; ++kt) {
      v16bf bmat = *(const v16bf*)(wt + (size_t)kt * 16384 +
                                        (size_t)nt * 512 + (size_t)lane * 16);
      acc = __builtin_amdgcn_wmma_f32_16x16x32_bf16(
          /*neg_a=*/false, a[kt].v, /*neg_b=*/false, bmat,
          /*c_mod=*/(short)0, acc, /*reuse_a=*/false, /*reuse_b=*/false);
    }
    // epilogue for this 16-column N-tile: tanh then dot with proj
#pragma unroll
    for (int r = 0; r < 8; ++r) rs[r] += tanhf(acc[r]) * pj;
  }

  // reduce across the 16 lanes of each half (columns) -> full dot per row
#pragma unroll
  for (int r = 0; r < 8; ++r) {
    float v = rs[r];
    v += __shfl_xor(v, 1);
    v += __shfl_xor(v, 2);
    v += __shfl_xor(v, 4);
    v += __shfl_xor(v, 8);
    rs[r] = v;
  }
  if (mloc == 0) {
#pragma unroll
    for (int r = 0; r < 8; ++r)
      s[m0 + half * 8 + r] = rs[r];   // lane0 -> rows 0..7, lane16 -> rows 8..15
  }
}

// ---------------------------------------------------------------------------
// Kernel 2: p[j,:] = softmax over B=8 of s[j,:], per timestep j.
// ---------------------------------------------------------------------------
__global__ void __launch_bounds__(256)
attn_softmaxB(const float* __restrict__ s, float* __restrict__ p) {
  int j = blockIdx.x * blockDim.x + threadIdx.x;
  if (j >= T_DIM) return;
  float v[B_DIM];
  float mx = -3.0e38f;
#pragma unroll
  for (int b = 0; b < B_DIM; ++b) { v[b] = s[j * B_DIM + b]; mx = fmaxf(mx, v[b]); }
  float sum = 0.0f;
#pragma unroll
  for (int b = 0; b < B_DIM; ++b) { v[b] = expf(v[b] - mx); sum += v[b]; }
  float inv = 1.0f / sum;
#pragma unroll
  for (int b = 0; b < B_DIM; ++b) p[j * B_DIM + b] = v[b] * inv;
}

// ---------------------------------------------------------------------------
// Kernel 3: out[i,b,d] = x[i,b,d]                        for i < W
//           out[i,b,d] = sum_{j=i-W}^{i-1} p[j,b]x[j,b,d] for i >= W
// Sliding-window sum, re-initialized per 32-step chunk (no long-range drift).
// grid: 32 chunks x 8 b x 2 d-halves; threads cover d (coalesced).
// ---------------------------------------------------------------------------
__global__ void __launch_bounds__(256)
attn_attend(const float* __restrict__ x,
            const float* __restrict__ p,
            float* __restrict__ out) {
  int c   = blockIdx.x >> 4;                 // 0..31 time chunk
  int rem = blockIdx.x & 15;
  int b   = rem >> 1;                        // 0..7
  int d   = ((rem & 1) << 8) + threadIdx.x;  // 0..511
  size_t bd = (size_t)b * D_DIM + d;
  const int i0 = c * W_WIN;

  if (c == 0) {                              // passthrough: out[0:W] = x[0:W]
#pragma unroll 4
    for (int k = 0; k < W_WIN; ++k)
      out[(size_t)k * (B_DIM * D_DIM) + bd] = x[(size_t)k * (B_DIM * D_DIM) + bd];
    return;
  }

  float S = 0.0f;
  for (int j = i0 - W_WIN; j < i0; ++j)
    S += p[j * B_DIM + b] * x[(size_t)j * (B_DIM * D_DIM) + bd];

  for (int k = 0; k < W_WIN; ++k) {
    int i = i0 + k;
    out[(size_t)i * (B_DIM * D_DIM) + bd] = S;
    S += p[i * B_DIM + b]            * x[(size_t)i * (B_DIM * D_DIM) + bd]
       - p[(i - W_WIN) * B_DIM + b]  * x[(size_t)(i - W_WIN) * (B_DIM * D_DIM) + bd];
  }
}

// ---------------------------------------------------------------------------
extern "C" void kernel_launch(void* const* d_in, const int* in_sizes, int n_in,
                              void* d_out, int out_size, void* d_ws, size_t ws_size,
                              hipStream_t stream) {
  const float* x    = (const float*)d_in[0];  // [1024, 8, 512]
  const float* wW   = (const float*)d_in[1];  // [512, 512]
  const float* proj = (const float*)d_in[2];  // [512, 1]
  // d_in[3] = attention_width (=32, baked into W_WIN)

  char* ws = (char*)d_ws;
  unsigned short* wbf = (unsigned short*)ws;                 // 512 KB bf16 W tiles
  float* s = (float*)(ws + (size_t)512 * 1024);              //  32 KB scores
  float* p = (float*)(ws + (size_t)512 * 1024 + 32 * 1024);  //  32 KB softmax
  float* out = (float*)d_out;

  attn_wconvert<<<64, 256, 0, stream>>>(wW, wbf);         // 16384 threads
  attn_scores  <<<64, 256, 0, stream>>>(x, wbf, proj, s); // 512 waves, 16 rows each
  attn_softmaxB<<<4, 256, 0, stream>>>(s, p);             // 1024 timesteps
  attn_attend  <<<512, 256, 0, stream>>>(x, p, out);      // 32 chunks x 8 b x 2 dhalf
}
